// WERSAAttention_11330123727384
// MI455X (gfx1250) — matrix operand order repaired and checked by reading, not compiled, not run on hardware
//
#include <hip/hip_runtime.h>
#include <cstdint>
#include <cstddef>

// ---------------------------------------------------------------------------
// Types for CDNA5 WMMA (wave32): v_wmma_f32_16x16x32_bf16
// ---------------------------------------------------------------------------
typedef __bf16 bf16;
typedef __attribute__((ext_vector_type(16))) __bf16 v16bf;
typedef __attribute__((ext_vector_type(8)))  float  v8f;
typedef __attribute__((ext_vector_type(4)))  unsigned int v4u;
typedef __attribute__((ext_vector_type(8)))  int  v8i;
typedef __attribute__((ext_vector_type(4)))  int  v4i;

union Frag16 {
    v16bf v;
    bf16  e[16];
    unsigned int d[8];
};

#if __has_builtin(__builtin_amdgcn_tensor_load_to_lds) && __has_builtin(__builtin_amdgcn_s_wait_tensorcnt)
#define HAVE_TDM 1
#else
#define HAVE_TDM 0
#endif

// Problem constants (from the reference)
#define H_   16
#define S_   4096
#define HD_  64
#define D_   1024
#define R_   1024
#define DQ_  256          // D/4
#define NBLK_ 64          // S / 64 blocks for denom
#define SQRT1_2 0.70710678118654752440f

// K-chunk and padded LDS row stride (dwords) for the N=64 LDS-staged GEMM
#define KC_   512
#define BROW_ 257         // KC_/2 dwords + 1 pad dword (bank-conflict-free)

// Escape the LDS pointer into opaque code that may write through it, so the
// compiler cannot conclude the LDS array is never stored (TDM writes it
// behind the compiler's back via descriptor-encoded addresses).
__device__ __forceinline__ void lds_escape(unsigned int* p) {
    asm volatile("" : "+v"(p) : : "memory");
}

// ===========================================================================
// N=64 per-head GEMM with B panel staged in LDS via the Tensor Data Mover.
// Double-buffered K-chunks of 512; TDM pad (1 dword / 256 dwords) gives a
// 257-dword LDS row stride => conflict-reduced B fragment reads.
// grid = (M/128, H), block = 256 (8 waves, one 16x64 strip each).
// ===========================================================================
__device__ __forceinline__ void tdm_stage_panel(const bf16* __restrict__ Bh,
                                                int ldb, int K, int chunk, int buf)
{
#if HAVE_TDM
    if ((threadIdx.x >> 5) == 0) {   // one wave issues the DMA
        const unsigned long long ga = (unsigned long long)(Bh + (long)chunk * KC_);
        v4u g0;
        g0[0] = 1u;                                          // count = 1
        g0[1] = (unsigned)(buf * 64 * BROW_ * 4);            // lds_addr (bytes)
        g0[2] = (unsigned)(ga & 0xffffffffull);              // global_addr lo
        g0[3] = (unsigned)((ga >> 32) & 0x01ffffffull) | (2u << 30);  // hi + type=2
        const unsigned d0 = (1u << 16)      // data_size = 1 -> 2 bytes
                          | (1u << 20)      // pad_enable
                          | (7u << 22)      // pad_interval code 7 = 256 dwords
                          | (0u << 25);     // pad_amount  code 0 = 1 dword
        const unsigned td0 = (unsigned)K;   // tensor_dim0
        const unsigned td1 = 64u;           // tensor_dim1
        v8i g1;
        g1[0] = (int)d0;
        g1[1] = (int)((td0 & 0xffffu) << 16);                 // [63:48] = td0 lo16
        g1[2] = (int)((td0 >> 16) | ((td1 & 0xffffu) << 16)); // td0 hi16 | td1 lo16
        g1[3] = (int)((td1 >> 16) | ((unsigned)KC_ << 16));   // td1 hi16 | tile_dim0
        g1[4] = (int)64;                                      // tile_dim1=64, tile_dim2=0
        g1[5] = (int)(unsigned)ldb;                           // tensor_dim0_stride lo32
        g1[6] = 0;                                            // stride hi16 | stride1 lo16
        g1[7] = 0;
        v4i g2; g2[0] = 1; g2[1] = 1; g2[2] = 0; g2[3] = 0;   // dims 2/3 = 1, tile_dim3=0
        v4i g3; g3[0] = 0; g3[1] = 0; g3[2] = 0; g3[3] = 0;
#if __clang_major__ >= 23
        v8i g4; g4[0]=0; g4[1]=0; g4[2]=0; g4[3]=0; g4[4]=0; g4[5]=0; g4[6]=0; g4[7]=0;
        __builtin_amdgcn_tensor_load_to_lds(g0, g1, g2, g3, g4, 0);
#else
        __builtin_amdgcn_tensor_load_to_lds(g0, g1, g2, g3, 0);
#endif
    }
#else
    (void)Bh; (void)ldb; (void)K; (void)chunk; (void)buf;
#endif
}

__global__ __launch_bounds__(256) void gemm_lds_n64(
    const bf16* __restrict__ A, long A_hs, int lda,
    const bf16* __restrict__ Bt, long B_hs, int ldb, int K,
    float* __restrict__ outf, long of_hs)
{
    __shared__ unsigned int Bp[2 * 64 * BROW_];   // 131,584 B

    const int lane = threadIdx.x & 31;
    const int wave = threadIdx.x >> 5;
    const int row  = lane & 15;
    const int hi   = lane >> 4;
    const int h    = blockIdx.y;
    const bf16* Ah = A  + (long)h * A_hs;
    const bf16* Bh = Bt + (long)h * B_hs;
    const int tm   = blockIdx.x * 8 + wave;
    const bf16* arow = Ah + (long)(tm * 16 + row) * lda;
    const int nch  = K / KC_;

#if !HAVE_TDM
    auto copy_panel = [&](int chunk, int buf) {
        const unsigned int* src = (const unsigned int*)Bh;
        const int kdw = chunk * (KC_ / 2);
        for (int i = threadIdx.x; i < 64 * (KC_ / 2); i += 256) {
            const int r = i >> 8, col = i & 255;
            Bp[buf * 64 * BROW_ + r * BROW_ + col] = src[(long)r * (ldb >> 1) + kdw + col];
        }
    };
#endif

    // stage chunk 0
#if HAVE_TDM
    tdm_stage_panel(Bh, ldb, K, 0, 0);
    if (wave == 0) __builtin_amdgcn_s_wait_tensorcnt(0);
    // Real (harmless) store into Bp: pad dword of row 0 (TDM skips pad dwords,
    // compute never reads them). Defeats "never stored => undef" folding.
    if (threadIdx.x == 0) ((volatile unsigned int*)Bp)[256] = 0u;
    lds_escape(&Bp[0]);
#else
    copy_panel(0, 0);
#endif
    __syncthreads();

    v8f c0 = {0.f,0.f,0.f,0.f,0.f,0.f,0.f,0.f};
    v8f c1 = c0, c2 = c0, c3 = c0;

    for (int c = 0; c < nch; ++c) {
        const int have_next = (c + 1 < nch);
#if HAVE_TDM
        if (have_next) tdm_stage_panel(Bh, ldb, K, c + 1, (c + 1) & 1);
#else
        if (have_next) copy_panel(c + 1, (c + 1) & 1);
#endif
        const int bufo = (c & 1) * 64 * BROW_;
        for (int k0 = 0; k0 < KC_; k0 += 32) {
            Frag16 a;
#pragma unroll
            for (int g = 0; g < 8; ++g) {
                const int ka = c * KC_ + k0 + ((g >> 2) << 4) + ((g & 3) << 1) + (hi << 3);
                a.d[g] = *(const unsigned int*)(arow + ka);
            }
            if (c * KC_ + k0 + 64 < K) __builtin_prefetch(arow + c * KC_ + k0 + 64, 0, 3);
#pragma unroll
            for (int t = 0; t < 4; ++t) {
                Frag16 b;
#pragma unroll
                for (int g = 0; g < 8; ++g) {
                    const int kb = k0 + ((g >> 2) << 4) + ((g & 3) << 1) + (hi << 3);
                    b.d[g] = Bp[bufo + (t * 16 + row) * BROW_ + (kb >> 1)];
                }
                v8f& acc = (t == 0) ? c0 : (t == 1) ? c1 : (t == 2) ? c2 : c3;
                acc = __builtin_amdgcn_wmma_f32_16x16x32_bf16(false, a.v, false, b.v,
                                                              (short)0, acc, false, false);
            }
        }
#if HAVE_TDM
        if (have_next) {
            if (wave == 0) __builtin_amdgcn_s_wait_tensorcnt(0);
            if (threadIdx.x == 0)
                ((volatile unsigned int*)Bp)[((c + 1) & 1) * 64 * BROW_ + 256] = 0u;
            lds_escape(&Bp[0]);
        }
#endif
        __syncthreads();
    }

#pragma unroll
    for (int t = 0; t < 4; ++t) {
        const v8f& acc = (t == 0) ? c0 : (t == 1) ? c1 : (t == 2) ? c2 : c3;
        const int n = t * 16 + row;
#pragma unroll
        for (int j = 0; j < 8; ++j) {
            const int m = tm * 16 + j + (hi << 3);
            outf[(long)h * of_hs + (long)m * 64 + n] = acc[j];
        }
    }
}

// ===========================================================================
// Generic bf16 GEMM:  C(M x N) = epilogue( A(MxK,row-major) * Bt(NxK)^T )
// Each wave computes a 16x64 strip: A fragment reused across 4 n-tiles.
// ===========================================================================
__global__ __launch_bounds__(256) void gemm_bf16(
    const bf16* __restrict__ A,  long A_hs, int lda,
    const bf16* __restrict__ Bt, long B_hs, int ldb,
    int M, int N, int K,
    const float* __restrict__ bias,   // length N, or nullptr
    float alpha, int relu,
    bf16*  __restrict__ outb, long ob_hs, int ldob,   // bf16 row-major out (or null)
    float* __restrict__ outf, long of_hs, int ldof,   // fp32 row-major out (or null)
    bf16*  __restrict__ outT, int trS)                // transposed [h][n][s] out (or null)
{
    const int lane = threadIdx.x & 31;
    const int wave = threadIdx.x >> 5;
    const int tilesN4 = N >> 6;                       // strips of 4 n-tiles
    const int totalTiles = (M >> 4) * tilesN4;
    const int tile = blockIdx.x * (blockDim.x >> 5) + wave;
    if (tile >= totalTiles) return;                   // wave-uniform

    const int tm  = tile / tilesN4;
    const int tn4 = tile - tm * tilesN4;
    const int h   = blockIdx.y;
    A  += (long)h * A_hs;
    Bt += (long)h * B_hs;

    const int row = lane & 15;
    const int hi  = lane >> 4;
    const bf16* arow = A  + (long)(tm * 16 + row) * lda;
    const bf16* brow = Bt + (long)(tn4 * 64 + row) * ldb;

    v8f c0 = {0.f,0.f,0.f,0.f,0.f,0.f,0.f,0.f};
    v8f c1 = c0, c2 = c0, c3 = c0;

    for (int k0 = 0; k0 < K; k0 += 32) {
        Frag16 a;
#pragma unroll
        for (int g = 0; g < 8; ++g) {
            const int ka = k0 + ((g >> 2) << 4) + ((g & 3) << 1) + (hi << 3);
            a.d[g] = *(const unsigned int*)(arow + ka);
        }
        if (k0 + 64 < K) __builtin_prefetch(arow + k0 + 64, 0, 3);
#pragma unroll
        for (int t = 0; t < 4; ++t) {
            Frag16 b;
            const bf16* bt = brow + (long)(t * 16) * ldb;
#pragma unroll
            for (int g = 0; g < 8; ++g) {
                const int kb = k0 + ((g >> 2) << 4) + ((g & 3) << 1) + (hi << 3);
                b.d[g] = *(const unsigned int*)(bt + kb);
            }
            v8f& acc = (t == 0) ? c0 : (t == 1) ? c1 : (t == 2) ? c2 : c3;
            acc = __builtin_amdgcn_wmma_f32_16x16x32_bf16(false, a.v, false, b.v,
                                                          (short)0, acc, false, false);
        }
    }

#pragma unroll
    for (int t = 0; t < 4; ++t) {
        const v8f& acc = (t == 0) ? c0 : (t == 1) ? c1 : (t == 2) ? c2 : c3;
        const int n = tn4 * 64 + t * 16 + row;
        const float bv = bias ? bias[n] : 0.f;
#pragma unroll
        for (int j = 0; j < 8; ++j) {
            const int m = tm * 16 + j + (hi << 3);
            float val = acc[j] + bv;
            if (relu) val = val > 0.f ? val : 0.f;
            val *= alpha;
            if (outb) outb[(long)h * ob_hs + (long)m * ldob + n] = (bf16)val;
            if (outf) outf[(long)h * of_hs + (long)m * ldof + n] = val;
            if (outT) {
                const int hh = m / trS, s = m - hh * trS;
                outT[((long)hh * N + n) * trS + s] = (bf16)val;
            }
        }
    }
}

// ===========================================================================
// Depthwise conv1d (kernel 33, SAME, zero pad) over S per channel -> bf16
// ===========================================================================
__global__ __launch_bounds__(256) void conv_kernel(
    const float* __restrict__ fs,   // [S, D]
    const float* __restrict__ cw,   // [D, 1, 33]
    const float* __restrict__ cb,   // [D]
    bf16* __restrict__ localb)      // [S, D]
{
    const int tid = blockIdx.x * 256 + threadIdx.x;
    const int d = tid & (D_ - 1);
    const int s = tid >> 10;
    float acc = cb[d];
#pragma unroll 1
    for (int t = 0; t < 33; ++t) {
        const int ss = s + t - 16;
        if (ss >= 0 && ss < S_) acc += fs[(long)ss * D_ + d] * cw[d * 33 + t];
    }
    localb[(long)s * D_ + d] = (bf16)acc;
}

// ===========================================================================
// Small transpose/convert kernels (fp32 -> bf16)
// ===========================================================================
__global__ void cvt_w1t(const float* __restrict__ w1, bf16* __restrict__ w1t) {
    const int tid = blockIdx.x * 256 + threadIdx.x;
    const int k = tid & (D_ - 1), n = tid >> 10;
    w1t[(long)n * D_ + k] = (bf16)w1[(long)k * DQ_ + n];
}
__global__ void cvt_rT(const float* __restrict__ rm, bf16* __restrict__ rT) {
    const int tid = blockIdx.x * 256 + threadIdx.x;
    const int d = tid & (HD_ - 1), r = tid >> 6;
    rT[(long)r * HD_ + d] = (bf16)rm[(long)d * R_ + r];
}
__global__ void cvt_vT(const float* __restrict__ v, bf16* __restrict__ vT) {
    const int tid = blockIdx.x * 256 + threadIdx.x;
    const int d = tid & (HD_ - 1);
    const int s = (tid >> 6) & (S_ - 1);
    const int h = tid >> 18;
    vT[((long)h * HD_ + d) * S_ + s] = (bf16)v[tid];
}

// ===========================================================================
// filters = sigmoid(h1 @ w2 + b2)   (4096 x 3, K=256) -- tiny
// ===========================================================================
__global__ void filters_kernel(
    const bf16* __restrict__ h1,    // [S, DQ]
    const float* __restrict__ w2,   // [DQ, 3]
    const float* __restrict__ b2,   // [3]
    float* __restrict__ filters)    // [S, 3]
{
    const int s = blockIdx.x * 256 + threadIdx.x;
    float a0 = b2[0], a1 = b2[1], a2 = b2[2];
    const bf16* hr = h1 + (long)s * DQ_;
#pragma unroll 4
    for (int i = 0; i < DQ_; ++i) {
        const float hv = (float)hr[i];
        a0 += hv * w2[i * 3 + 0];
        a1 += hv * w2[i * 3 + 1];
        a2 += hv * w2[i * 3 + 2];
    }
    filters[s * 3 + 0] = 1.f / (1.f + __expf(-a0));
    filters[s * 3 + 1] = 1.f / (1.f + __expf(-a1));
    filters[s * 3 + 2] = 1.f / (1.f + __expf(-a2));
}

// ===========================================================================
// GeLU + 2-level Haar DWT * filters + IDWT (per-4-samples linear map);
// output scaled by 1/bandwidth, stored bf16.
// ===========================================================================
__global__ __launch_bounds__(256) void wavelet_kernel(
    const float* __restrict__ x,       // [H, S, Hd] fp32
    bf16* __restrict__ out,            // [H, S, Hd] bf16
    const float* __restrict__ filters, // [S, 3]
    const float* __restrict__ sw,      // [3]
    const float* __restrict__ lbw)     // [1]
{
    const int tid = blockIdx.x * 256 + threadIdx.x;
    const int d   = tid & (HD_ - 1);
    const int grp = (tid >> 6) & ((S_ / 4) - 1);
    const int h   = tid >> 16;
    const int s0  = grp * 4;
    const long base = ((long)h * S_ + s0) * HD_ + d;

    float x0 = x[base], x1 = x[base + HD_], x2 = x[base + 2 * HD_], x3 = x[base + 3 * HD_];
    x0 = 0.5f * x0 * (1.f + erff(x0 * SQRT1_2));
    x1 = 0.5f * x1 * (1.f + erff(x1 * SQRT1_2));
    x2 = 0.5f * x2 * (1.f + erff(x2 * SQRT1_2));
    x3 = 0.5f * x3 * (1.f + erff(x3 * SQRT1_2));

    const float lb = lbw[0];
    const float bw = (fmaxf(lb, 0.f) + log1pf(__expf(-fabsf(lb)))) + 0.1f;
    const float inv = 1.f / bw;

    const float f1e = filters[s0 * 3 + 0] * sw[0];
    const float f1o = filters[(s0 + 2) * 3 + 0] * sw[0];
    const float f2  = filters[s0 * 3 + 1] * sw[1];
    const float fa  = filters[s0 * 3 + 2] * sw[2];

    const float d1e = (x0 - x1) * SQRT1_2 * f1e;
    const float d1o = (x2 - x3) * SQRT1_2 * f1o;
    const float a1e = (x0 + x1) * SQRT1_2;
    const float a1o = (x2 + x3) * SQRT1_2;
    const float d2  = (a1e - a1o) * SQRT1_2 * f2;
    const float a2  = (a1e + a1o) * SQRT1_2 * fa;

    const float c0 = (a2 + d2) * SQRT1_2;
    const float c1 = (a2 - d2) * SQRT1_2;

    out[base]           = (bf16)(((c0 + d1e) * SQRT1_2) * inv);
    out[base + HD_]     = (bf16)(((c0 - d1e) * SQRT1_2) * inv);
    out[base + 2 * HD_] = (bf16)(((c1 + d1o) * SQRT1_2) * inv);
    out[base + 3 * HD_] = (bf16)(((c1 - d1o) * SQRT1_2) * inv);
}

// ===========================================================================
// Per-(h,r) scan over S producing exclusive 64-row block prefixes of k_proj.
// ===========================================================================
__global__ void kscan_kernel(const bf16* __restrict__ kT, float* __restrict__ Kpre) {
    const int tid = blockIdx.x * 256 + threadIdx.x;   // H*R
    const int h = tid >> 10, r = tid & (R_ - 1);
    const bf16* p = kT + ((long)h * R_ + r) * S_;
    float run = 0.f;
    for (int blk = 0; blk < NBLK_; ++blk) {
        Kpre[((long)h * NBLK_ + blk) * R_ + r] = run;
        const bf16* pb = p + blk * 64;
#pragma unroll 8
        for (int i = 0; i < 64; ++i) run += (float)pb[i];
    }
}

// ===========================================================================
// denom: blocked 64x64 Gram (WMMA) + q.Kpre GEMV + masked row-sum (+1e-6).
// ===========================================================================
__global__ __launch_bounds__(256) void gram_denom(
    const bf16* __restrict__ qproj,  // [H*S, R]
    const bf16* __restrict__ kT,     // [H, R, S]
    const float* __restrict__ Kpre,  // [H, NBLK, R]
    float* __restrict__ denom)       // [H*S]
{
    __shared__ float G[64 * 64];
    __shared__ float qd[4 * 64];
    const int h = blockIdx.y, blk = blockIdx.x;
    const int lane = threadIdx.x & 31, wave = threadIdx.x >> 5;
    const int row = lane & 15, hi = lane >> 4;

    const bf16* Aq = qproj + ((long)h * S_ + blk * 64) * R_;
    const bf16* Bk = kT + (long)h * R_ * S_ + blk * 64;

    for (int rep = 0; rep < 2; ++rep) {
        const int tile = wave + rep * 8;
        const int tm = tile >> 2, tn = tile & 3;
        const bf16* arow = Aq + (long)(tm * 16 + row) * R_;
        const bf16* bcol = Bk + tn * 16 + row;
        v8f c = {0.f,0.f,0.f,0.f,0.f,0.f,0.f,0.f};
        for (int k0 = 0; k0 < R_; k0 += 32) {
            Frag16 a, b;
#pragma unroll
            for (int g = 0; g < 8; ++g) {
                const int ka = k0 + ((g >> 2) << 4) + ((g & 3) << 1) + (hi << 3);
                a.d[g] = *(const unsigned int*)(arow + ka);
                const int kb = k0 + ((g >> 2) << 4) + ((g & 3) << 1) + (hi << 3);
                b.e[2 * g]     = bcol[(long)kb * S_];
                b.e[2 * g + 1] = bcol[(long)(kb + 1) * S_];
            }
            c = __builtin_amdgcn_wmma_f32_16x16x32_bf16(false, a.v, false, b.v,
                                                        (short)0, c, false, false);
        }
#pragma unroll
        for (int j = 0; j < 8; ++j)
            G[(tm * 16 + j + (hi << 3)) * 64 + tn * 16 + row] = c[j];
    }
    __syncthreads();

    {
        const int sL = threadIdx.x & 63, q = threadIdx.x >> 6;
        const bf16*  qr = Aq + (long)sL * R_ + q * (R_ / 4);
        const float* pr = Kpre + ((long)h * NBLK_ + blk) * R_ + q * (R_ / 4);
        float acc = 0.f;
#pragma unroll 4
        for (int i = 0; i < R_ / 4; ++i) acc += (float)qr[i] * pr[i];
        qd[q * 64 + sL] = acc;
    }
    __syncthreads();

    if (threadIdx.x < 64) {
        const int s = threadIdx.x;
        float dacc = 1e-6f + qd[s] + qd[64 + s] + qd[128 + s] + qd[192 + s];
        for (int s2 = 0; s2 <= s; ++s2) dacc += G[s * 64 + s2];
        denom[(long)h * S_ + blk * 64 + s] = dacc;
    }
}

// ===========================================================================
// kv_cumsum over R (axis 2 of bhrd), stored transposed-bf16 [H][Hd][R]
// ===========================================================================
__global__ void kvcum_kernel(const float* __restrict__ kv, bf16* __restrict__ kvcT) {
    const int tid = blockIdx.x * 256 + threadIdx.x;
    if (tid >= H_ * HD_) return;
    const int h = tid >> 6, d = tid & (HD_ - 1);
    float run = 0.f;
    for (int r = 0; r < R_; ++r) {
        run += kv[((long)h * R_ + r) * HD_ + d];
        kvcT[((long)h * HD_ + d) * R_ + r] = (bf16)run;
    }
}

// ===========================================================================
// Final: divide by denom, layernorm over Hd, gamma/beta. One wave per row.
// ===========================================================================
__global__ __launch_bounds__(256) void ln_kernel(
    const float* __restrict__ attn_raw,
    const float* __restrict__ denom,
    const float* __restrict__ gamma, const float* __restrict__ beta,
    float* __restrict__ out)
{
    const int lane = threadIdx.x & 31;
    const int wave = threadIdx.x >> 5;
    const long rowid = (long)blockIdx.x * 8 + wave;
    const float inv = 1.f / denom[rowid];
    const float* ar = attn_raw + rowid * 64;
    const float a0 = ar[lane] * inv;
    const float a1 = ar[lane + 32] * inv;
    float sum = a0 + a1, sq = a0 * a0 + a1 * a1;
#pragma unroll
    for (int off = 16; off >= 1; off >>= 1) {
        sum += __shfl_xor(sum, off, 32);
        sq  += __shfl_xor(sq,  off, 32);
    }
    const float mu   = sum * (1.f / 64.f);
    const float var  = sq * (1.f / 64.f) - mu * mu;
    const float rstd = rsqrtf(var + 1e-5f);
    float* orow = out + rowid * 64;
    orow[lane]      = (a0 - mu) * rstd * gamma[lane]      + beta[lane];
    orow[lane + 32] = (a1 - mu) * rstd * gamma[lane + 32] + beta[lane + 32];
}

// ===========================================================================
// Host launcher
// ===========================================================================
extern "C" void kernel_launch(void* const* d_in, const int* in_sizes, int n_in,
                              void* d_out, int out_size, void* d_ws, size_t ws_size,
                              hipStream_t stream) {
    const float* q    = (const float*)d_in[0];
    const float* k    = (const float*)d_in[1];
    const float* v    = (const float*)d_in[2];
    const float* fs   = (const float*)d_in[3];
    const float* cw   = (const float*)d_in[4];
    const float* cb   = (const float*)d_in[5];
    const float* w1   = (const float*)d_in[6];
    const float* b1   = (const float*)d_in[7];
    const float* w2   = (const float*)d_in[8];
    const float* b2   = (const float*)d_in[9];
    const float* lbw  = (const float*)d_in[10];
    const float* sw   = (const float*)d_in[11];
    const float* rq   = (const float*)d_in[12];
    const float* rk   = (const float*)d_in[13];
    const float* lng  = (const float*)d_in[14];
    const float* lnb  = (const float*)d_in[15];
    float* out = (float*)d_out;

    size_t off = 0;
    auto alloc = [&](size_t bytes) -> void* {
        void* p = (char*)d_ws + off;
        off += (bytes + 255) & ~(size_t)255;
        return p;
    };
    bf16*  local_b  = (bf16*) alloc((size_t)S_ * D_ * 2);
    bf16*  h1       = (bf16*) alloc((size_t)S_ * DQ_ * 2);
    bf16*  w1t      = (bf16*) alloc((size_t)DQ_ * D_ * 2);
    float* filters  = (float*)alloc((size_t)S_ * 3 * 4);
    bf16*  qfb      = (bf16*) alloc((size_t)H_ * S_ * HD_ * 2);
    bf16*  kfb      = (bf16*) alloc((size_t)H_ * S_ * HD_ * 2);
    bf16*  rqT      = (bf16*) alloc((size_t)R_ * HD_ * 2);
    bf16*  rkT      = (bf16*) alloc((size_t)R_ * HD_ * 2);
    bf16*  vT       = (bf16*) alloc((size_t)H_ * HD_ * S_ * 2);
    bf16*  qproj    = (bf16*) alloc((size_t)H_ * S_ * R_ * 2);
    bf16*  kT       = (bf16*) alloc((size_t)H_ * R_ * S_ * 2);
    float* Kpre     = (float*)alloc((size_t)H_ * NBLK_ * R_ * 4);
    float* kv       = (float*)alloc((size_t)H_ * R_ * HD_ * 4);
    bf16*  kvcT     = (bf16*) alloc((size_t)H_ * HD_ * R_ * 2);
    float* denom    = (float*)alloc((size_t)H_ * S_ * 4);
    float* attn_raw = (float*)alloc((size_t)H_ * S_ * HD_ * 4);

    // ---- filter branch ----
    conv_kernel<<<(S_ * D_) / 256, 256, 0, stream>>>(fs, cw, cb, local_b);
    cvt_w1t<<<(DQ_ * D_) / 256, 256, 0, stream>>>(w1, w1t);
    gemm_bf16<<<dim3((S_ / 16) * (DQ_ / 64) / 8, 1), 256, 0, stream>>>(
        local_b, 0, D_, w1t, 0, D_, S_, DQ_, D_,
        b1, 1.f, 1, h1, 0, DQ_, nullptr, 0, 0, nullptr, 1);
    filters_kernel<<<S_ / 256, 256, 0, stream>>>(h1, w2, b2, filters);

    // ---- wavelet-filtered q, k ----
    wavelet_kernel<<<(H_ * (S_ / 4) * HD_) / 256, 256, 0, stream>>>(q, qfb, filters, sw, lbw);
    wavelet_kernel<<<(H_ * (S_ / 4) * HD_) / 256, 256, 0, stream>>>(k, kfb, filters, sw, lbw);

    // ---- random feature projections ----
    cvt_rT<<<(R_ * HD_) / 256, 256, 0, stream>>>(rq, rqT);
    cvt_rT<<<(R_ * HD_) / 256, 256, 0, stream>>>(rk, rkT);
    cvt_vT<<<(H_ * S_ * HD_) / 256, 256, 0, stream>>>(v, vT);

    gemm_bf16<<<dim3(((H_ * S_) / 16) * (R_ / 64) / 8, 1), 256, 0, stream>>>(
        qfb, 0, HD_, rqT, 0, HD_, H_ * S_, R_, HD_,
        nullptr, 1.f / 32.f, 1, qproj, 0, R_, nullptr, 0, 0, nullptr, 1);
    gemm_bf16<<<dim3(((H_ * S_) / 16) * (R_ / 64) / 8, 1), 256, 0, stream>>>(
        kfb, 0, HD_, rkT, 0, HD_, H_ * S_, R_, HD_,
        nullptr, 1.f / 32.f, 1, nullptr, 0, 0, nullptr, 0, 0, kT, S_);

    // ---- denom ----
    kscan_kernel<<<(H_ * R_) / 256, 256, 0, stream>>>(kT, Kpre);
    gram_denom<<<dim3(NBLK_, H_), 256, 0, stream>>>(qproj, kT, Kpre, denom);

    // ---- kv = k_proj^T @ v per head (LDS/TDM-staged B panel) ----
    gemm_lds_n64<<<dim3(R_ / 128, H_), 256, 0, stream>>>(
        kT, (long)R_ * S_, S_, vT, (long)HD_ * S_, S_, S_, kv, (long)R_ * HD_);
    kvcum_kernel<<<(H_ * HD_ + 255) / 256, 256, 0, stream>>>(kv, kvcT);

    // ---- attn_raw = q_proj @ kv_cumsum per head (LDS/TDM-staged B panel) ----
    gemm_lds_n64<<<dim3(S_ / 128, H_), 256, 0, stream>>>(
        qproj, (long)S_ * R_, R_, kvcT, (long)HD_ * R_, R_, R_, attn_raw, (long)S_ * HD_);

    // ---- divide by denom + layernorm ----
    ln_kernel<<<(H_ * S_) / 8, 256, 0, stream>>>(attn_raw, denom, lng, lnb, out);

    (void)in_sizes; (void)n_in; (void)out_size; (void)ws_size;
}
